// Construct_76484777607483
// MI455X (gfx1250) — matrix operand order repaired and checked by reading, not compiled, not run on hardware
//
#include <hip/hip_runtime.h>

typedef __attribute__((ext_vector_type(16))) __bf16 v16bf;
typedef __attribute__((ext_vector_type(8)))  __bf16 v8bf;
typedef __attribute__((ext_vector_type(8)))  float  v8f;

typedef __attribute__((ext_vector_type(4))) unsigned int u32x4;
typedef __attribute__((ext_vector_type(8))) int          i32x8;
typedef __attribute__((ext_vector_type(4))) int          i32x4;

#define C_CH 256
#define HH 64
#define WW 256
#define PIX_PER_PLANE (HH * WW)     // 16384
#define HOUT (12 * HH)              // 768

// LDS map: [0,128K) = swizzled W hi (bf16), [128K,256K) = swizzled W lo,
// [256K,257K) = bias f32. 257 KB fits CDNA5's 320 KB/WGP LDS.
#define LDS_LO_OFF   (128 * 1024)
#define LDS_BIAS_OFF (256 * 1024)
#define LDS_BYTES    (256 * 1024 + 1024)

struct XPtrs { const float* p[12]; };

// ---------------------------------------------------------------------------
// Prep: swizzle W[c][o] (f32) into per-lane WMMA A-fragment order, split into
// bf16 hi + lo residual images (65536 bf16 each) in d_ws.
// Position encoding (bf16 units):
//   pos = ((((ot*8 + chunk)*2 + q)*32 + lane)*8 + e)
// A-fragment element j = q*8 + e maps to K = q*16 + half*8 + e (ISA 16-bit A
// 16x32 layout), row m = lane&15. So c = chunk*32 + q*16 + half*8 + e,
// o = ot*16 + m. The q-split keeps each ds_load_b128 bank-conflict free.
// ---------------------------------------------------------------------------
__global__ void prep_w_kernel(const float* __restrict__ W,
                              __bf16* __restrict__ whi,
                              __bf16* __restrict__ wlo) {
  int pos   = blockIdx.x * 256 + threadIdx.x;   // 0..65535
  int e     =  pos        & 7;
  int lane  = (pos >> 3)  & 31;
  int q     = (pos >> 8)  & 1;
  int chunk = (pos >> 9)  & 7;
  int ot    =  pos >> 12;
  int half  = lane >> 4;
  int m     = lane & 15;
  int c = chunk * 32 + q * 16 + half * 8 + e;
  int o = ot * 16 + m;
  float w   = W[c * C_CH + o];
  __bf16 hi = (__bf16)w;
  __bf16 lo = (__bf16)(w - (float)hi);
  whi[pos] = hi;
  wlo[pos] = lo;
}

// ---------------------------------------------------------------------------
// Main: per block = (tensor i, batch b, 512-pixel segment). 8 waves.
// W staging: one TDM tensor_load_to_lds (256KB flat copy, wave 0 issues,
// TENSORcnt wait, then workgroup barrier publishes LDS).
// Each wave: 16-pixel tile -> load X fragments once (f32 -> bf16 hi/lo),
// then 16 output-channel tiles x 8 K-chunks x 3 bf16 WMMAs (error-compensated
// hi/lo product), fused bias*count at store.
// ---------------------------------------------------------------------------
__global__ __launch_bounds__(256)
void conv1x1_wmma_kernel(XPtrs xs,
                         const __bf16* __restrict__ wswz,   // hi||lo, 256KB
                         const float*  __restrict__ bias,
                         float* __restrict__ out) {
  extern __shared__ char smem[];

#if defined(__gfx1250__) && __has_builtin(__builtin_amdgcn_tensor_load_to_lds)
  // ---- TDM: DMA the 256KB swizzled-W image (d_ws) into LDS[0..256K) -------
  // Descriptor per CDNA5 ISA ch.8 (Tensor DMA Descriptor):
  //   G0: count=1, lds_addr=0, global_addr=wswz, type=2
  //   G1: data_size=3 (8B), tensor_dim0=tile_dim0=stride0=32768 (=256KB),
  //       tensor_dim1=tile_dim1=1, workgroup_mask=0 (not in cluster)
  //   G2/G3: zero (dims 2..4 unused)
  if (threadIdx.x < 32) {                       // wave 0 only (TDM ignores EXEC)
    unsigned long long ga = (unsigned long long)(const void*)wswz;
    u32x4 g0;
    g0[0] = 1u;                                  // count=1, user descriptor
    g0[1] = 0u;                                  // lds_addr = 0
    g0[2] = (unsigned int)(ga & 0xFFFFFFFFu);    // global_addr[31:0]
    g0[3] = (unsigned int)((ga >> 32) & 0x01FFFFFFu) | (2u << 30); // [56:32]|type=2
    i32x8 g1;
    g1[0] = 0x00030000;                          // data_size=3 (8B), mask=0
    g1[1] = (int)0x80000000u;                    // tensor_dim0[15:0]=0x8000 @[31:16]
    g1[2] = 0x00010000;                          // tensor_dim1=1 @[31:16]
    g1[3] = (int)0x80000000u;                    // tile_dim0=0x8000 @[31:16]
    g1[4] = 0x00000001;                          // tile_dim1=1, tile_dim2=0
    g1[5] = 0x00008000;                          // tensor_dim0_stride=32768
    g1[6] = 0;
    g1[7] = 0;
    i32x4 z4 = {0, 0, 0, 0};
#if __clang_major__ >= 23
    i32x8 z8 = {0, 0, 0, 0, 0, 0, 0, 0};
    __builtin_amdgcn_tensor_load_to_lds(g0, g1, z4, z4, z8, 0);
#else
    __builtin_amdgcn_tensor_load_to_lds(g0, g1, z4, z4, 0);
#endif
    __builtin_amdgcn_s_wait_tensorcnt(0);        // wave 0 owns TENSORcnt
  }
  ((float*)(smem + LDS_BIAS_OFF))[threadIdx.x] = bias[threadIdx.x];
  __syncthreads();                               // publish LDS to all 8 waves
#else
  // ---- Fallback: cooperative 16B-per-lane copy loop -----------------------
  {
    const uint4* src = (const uint4*)wswz;
    uint4*       dst = (uint4*)smem;
    #pragma unroll 4
    for (int it = 0; it < 64; ++it) {
      int idx = it * 256 + (int)threadIdx.x;
      dst[idx] = src[idx];
    }
    ((float*)(smem + LDS_BIAS_OFF))[threadIdx.x] = bias[threadIdx.x];
  }
  __syncthreads();
#endif

  const int lane = threadIdx.x & 31;
  const int wid  = threadIdx.x >> 5;
  const int half = lane >> 4;
  const int n    = lane & 15;        // B/D column within tile

  const int seg = blockIdx.x & 31;          // 32 segments of 512 pixels
  const int b   = (blockIdx.x >> 5) & 1;    // batch
  const int i   = blockIdx.x >> 6;          // source tensor 0..11

  const float* xb   = xs.p[i] + (size_t)b * C_CH * PIX_PER_PLANE;
  float*       outb = out     + (size_t)b * C_CH * HOUT * WW;

  for (int g = 0; g < 4; ++g) {
    const int pix0 = seg * 512 + g * 128 + wid * 16;   // 16 px, same row h
    const int h  = pix0 >> 8;
    const int w0 = pix0 & 255;
    const int r  = 12 * h + i;                          // interleaved out row
    const int cmin = (r < 11) ? r : 11;
    const int csub = (r > 756) ? (r - 756) : 0;
    const float countf = (float)(cmin - csub + 1);      // bias multiplicity

    // ---- Load B fragments (X tile), reused across all 16 o-tiles ----------
    // B element j (32x16 bf16 layout): K = (j>>3)*16 + half*8 + (j&7), column = n.
    const float* xp = xb + (size_t)pix0 + n;
    v16bf Bh[8], Bl[8];
    #pragma unroll
    for (int ch = 0; ch < 8; ++ch) {
      #pragma unroll
      for (int j = 0; j < 16; ++j) {
        int c   = ch * 32 + (j >> 3) * 16 + half * 8 + (j & 7);
        float v = xp[(size_t)c * PIX_PER_PLANE];
        __bf16 hv = (__bf16)v;
        Bh[ch][j] = hv;
        Bl[ch][j] = (__bf16)(v - (float)hv);
      }
    }

    float* op = outb + (size_t)r * WW + w0 + n;

    for (int ot = 0; ot < 16; ++ot) {
      v8f acc = {0.f, 0.f, 0.f, 0.f, 0.f, 0.f, 0.f, 0.f};
      #pragma unroll
      for (int ch = 0; ch < 8; ++ch) {
        const int fb = (((ot * 8 + ch) * 2) * 32 + lane) * 16;  // bytes, q=0
        v8bf a0 = *(const v8bf*)(smem + fb);
        v8bf a1 = *(const v8bf*)(smem + fb + 512);               // q=1
        v8bf l0 = *(const v8bf*)(smem + LDS_LO_OFF + fb);
        v8bf l1 = *(const v8bf*)(smem + LDS_LO_OFF + fb + 512);
        v16bf Ah, Al;
        #pragma unroll
        for (int t = 0; t < 8; ++t) {
          Ah[t] = a0[t]; Ah[8 + t] = a1[t];
          Al[t] = l0[t]; Al[8 + t] = l1[t];
        }
        // (Ah+Al)(Bh+Bl) ~= Ah*Bh + Ah*Bl + Al*Bh  (drop 2^-16 term)
        acc = __builtin_amdgcn_wmma_f32_16x16x32_bf16(false, Ah, false, Bh[ch],
                                                      (short)0, acc, false, false);
        acc = __builtin_amdgcn_wmma_f32_16x16x32_bf16(false, Ah, false, Bl[ch],
                                                      (short)0, acc, false, false);
        acc = __builtin_amdgcn_wmma_f32_16x16x32_bf16(false, Al, false, Bh[ch],
                                                      (short)0, acc, false, false);
      }
      // D layout: VGPR rr -> M = rr + 8*half, column = n. M is output channel.
      #pragma unroll
      for (int rr = 0; rr < 8; ++rr) {
        int o    = ot * 16 + half * 8 + rr;
        float bv = ((const float*)(smem + LDS_BIAS_OFF))[o];
        op[(size_t)o * (HOUT * WW)] = acc[rr] + bv * countf;
      }
    }
  }
}

extern "C" void kernel_launch(void* const* d_in, const int* in_sizes, int n_in,
                              void* d_out, int out_size, void* d_ws, size_t ws_size,
                              hipStream_t stream) {
  (void)in_sizes; (void)n_in; (void)out_size; (void)ws_size;
  XPtrs xs;
  for (int k = 0; k < 12; ++k) xs.p[k] = (const float*)d_in[k];
  const float* W    = (const float*)d_in[12];
  const float* bias = (const float*)d_in[13];

  __bf16* whi = (__bf16*)d_ws;           // 65536 bf16 (128KB)
  __bf16* wlo = whi + 65536;             // 65536 bf16 (128KB)

  prep_w_kernel<<<256, 256, 0, stream>>>(W, whi, wlo);

  // grid: 12 tensors x 2 batches x 32 segments = 768 blocks, 8 waves each.
  conv1x1_wmma_kernel<<<768, 256, LDS_BYTES, stream>>>(xs, whi, bias,
                                                       (float*)d_out);
}
